// SmallDataDecoderViT_2482491097851
// MI455X (gfx1250) — matrix operand order repaired and checked by reading, not compiled, not run on hardware
//
#include <hip/hip_runtime.h>
#include <hip/hip_bf16.h>
#include <math.h>

typedef __attribute__((ext_vector_type(16))) _Float16 v16h;
typedef __attribute__((ext_vector_type(8)))  _Float16 v8h;
typedef __attribute__((ext_vector_type(8)))  float    v8f;

#define DEV __device__ __forceinline__

// ---------------- geometry ----------------
constexpr int GHN   = 29;
constexpr int NTOK  = GHN * GHN;       // 841 tokens
constexpr int BATCH = 16;
constexpr int NT    = BATCH * NTOK;    // 13456 rows
constexpr int E     = 384;
constexpr int NH    = 6;
constexpr int HD    = 64;
constexpr int DEPTH = 8;
constexpr int TOKD  = 3840;
constexpr int HIDD  = 768;
constexpr int FFD   = 1536;
constexpr int NPAD  = 864;             // keys padded to 27*32
constexpr int QTILES= 53;              // ceil(841/16)
constexpr int KTILES= 27;              // 864/32
constexpr int QKVS  = BATCH * NH * NPAD * HD;  // elems per q/k/v tensor

// GEMM tiling
constexpr int BM = 128, BN = 128, BK = 32, LDT = BK + 8;

// ---------------- WMMA fragment loaders ----------------
// A fragment (16xK=32 f16): lane = M row; halves {0..7,16..23} (lanes<16) / {8..15,24..31}
DEV v16h load_frag_a(const _Float16* base, int stride, int lane) {
  const int r  = lane & 15;
  const int hi = (lane >> 4) & 1;
  const _Float16* p = base + (size_t)r * stride + hi * 8;
  v8h lo = *(const v8h*)p;
  v8h hi8 = *(const v8h*)(p + 16);
  v16h o;
  for (int i = 0; i < 8; ++i) { o[i] = lo[i]; o[i + 8] = hi8[i]; }
  return o;
}

// B fragment (K=32 x 16 f16) loaded from B^T stored [N][K]:
// lane = N col; halves {0..7,16..23} (lanes<16) / {8..15,24..31}, matching A
DEV v16h load_frag_b(const _Float16* baseT, int stride, int lane) {
  const int n  = lane & 15;
  const int hi = (lane >> 4) & 1;
  const _Float16* p = baseT + (size_t)n * stride + hi * 8;
  v8h a = *(const v8h*)p;
  v8h b = *(const v8h*)(p + 16);
  v16h o;
  for (int i = 0; i < 8; ++i) { o[i] = a[i]; o[i + 8] = b[i]; }
  return o;
}

DEV float gelu_exact(float x) {
  return 0.5f * x * (1.0f + erff(x * 0.70710678118654752f));
}

// CDNA5 async global->LDS tile fill (tracked by ASYNCcnt, no VGPR round trip)
DEV void async_fill_b128(const _Float16* gp, const _Float16* lp) {
  unsigned lds_off = (unsigned)(size_t)lp;
  asm volatile("global_load_async_to_lds_b128 %0, %1, off"
               :: "v"(lds_off), "v"(gp) : "memory");
}

// ---------------- patchify: 5 shifted patch embeddings fused ----------------
__global__ void patchify_k(const float* __restrict__ x, _Float16* __restrict__ p, long total) {
  long idx = (long)blockIdx.x * blockDim.x + threadIdx.x;
  if (idx >= total) return;
  int col = (int)(idx % TOKD);
  int row = (int)(idx / TOKD);
  int b = row / NTOK, n = row - b * NTOK;
  int gy = n / GHN, gx = n - gy * GHN;
  int s  = col / 768, r1 = col - s * 768;
  int c  = r1 >> 8,   r2 = r1 & 255;
  int py = r2 >> 4,   px = r2 & 15;
  const int dy[5] = {0, 8, -8, 0, 0};
  const int dx[5] = {0, 0, 0, 8, -8};
  int yy = gy * 16 + py - dy[s];
  int xx = gx * 16 + px - dx[s];
  float v = 0.f;
  if (yy >= 0 && yy < 464 && xx >= 0 && xx < 464)
    v = x[(((size_t)b * 3 + c) * 464 + yy) * 464 + xx];
  p[idx] = (_Float16)v;
}

// ---------------- LayerNorm over 3840 (in-place on f16 patches) ----------------
__global__ __launch_bounds__(256) void ln_tok_k(_Float16* __restrict__ p,
                                                const float* __restrict__ g,
                                                const float* __restrict__ beta) {
  const int row = blockIdx.x, tid = threadIdx.x;
  _Float16* pr = p + (size_t)row * TOKD;
  float vals[15], s1 = 0.f, s2 = 0.f;
  for (int j = 0; j < 15; ++j) {
    float v = (float)pr[tid + j * 256];
    vals[j] = v; s1 += v; s2 += v * v;
  }
  __shared__ float sh1[256], sh2[256];
  sh1[tid] = s1; sh2[tid] = s2; __syncthreads();
  for (int o = 128; o > 0; o >>= 1) {
    if (tid < o) { sh1[tid] += sh1[tid + o]; sh2[tid] += sh2[tid + o]; }
    __syncthreads();
  }
  float mean = sh1[0] * (1.f / TOKD);
  float var  = sh2[0] * (1.f / TOKD) - mean * mean;
  float rs   = rsqrtf(var + 1e-5f);
  for (int j = 0; j < 15; ++j) {
    int c = tid + j * 256;
    pr[c] = (_Float16)((vals[j] - mean) * rs * g[c] + beta[c]);
  }
}

// ---------------- LayerNorm over 384 (f32 residual -> f16) ----------------
__global__ __launch_bounds__(128) void ln384_k(const float* __restrict__ h,
                                               const float* __restrict__ g,
                                               const float* __restrict__ beta,
                                               _Float16* __restrict__ y) {
  const int row = blockIdx.x, tid = threadIdx.x;
  const float* hr = h + (size_t)row * E;
  float v[3], s1 = 0.f, s2 = 0.f;
  for (int j = 0; j < 3; ++j) { v[j] = hr[tid + j * 128]; s1 += v[j]; s2 += v[j] * v[j]; }
  __shared__ float sh1[128], sh2[128];
  sh1[tid] = s1; sh2[tid] = s2; __syncthreads();
  for (int o = 64; o > 0; o >>= 1) {
    if (tid < o) { sh1[tid] += sh1[tid + o]; sh2[tid] += sh2[tid + o]; }
    __syncthreads();
  }
  float mean = sh1[0] * (1.f / E);
  float var  = sh2[0] * (1.f / E) - mean * mean;
  float rs   = rsqrtf(var + 1e-5f);
  for (int j = 0; j < 3; ++j) {
    int c = tid + j * 128;
    y[(size_t)row * E + c] = (_Float16)((v[j] - mean) * rs * g[c] + beta[c]);
  }
}

// ---------------- weight transpose + f16 convert: wt[o][i] = w[i][o] ----------------
__global__ void wtrans_k(const float* __restrict__ w, _Float16* __restrict__ wt,
                         int In, int Out) {
  long idx = (long)blockIdx.x * blockDim.x + threadIdx.x;
  if (idx >= (long)In * Out) return;
  int o = (int)(idx / In), i = (int)(idx - (long)o * In);
  wt[idx] = (_Float16)w[(size_t)i * Out + o];
}

// ---------------- generic WMMA GEMM: C = A[M,K] * Bt[N,K]^T, fused epilogues ----------------
// 128x128 block tile, 8 waves each 32x64 (8 WMMAs/K-step), double-buffered
// async global->LDS staging. Epilogues:
// mode 0: out16 = gelu(acc + bias)           (f16)
// mode 1: out32 = acc + bias                 (f32)
// mode 3: hrmw += ls * (acc + bias)          (f32 residual RMW)
// mode 4: qkv scatter (no bias), out16 = qkv base
__global__ __launch_bounds__(256) void gemm_f16_wmma(
    const _Float16* __restrict__ A, int lda,
    const _Float16* __restrict__ Bt,
    int M, int N, int K, int mode,
    const float* __restrict__ bias, const float* __restrict__ ls,
    _Float16* __restrict__ out16, float* __restrict__ out32,
    float* __restrict__ hrmw) {
  __shared__ _Float16 As[2][BM][LDT];
  __shared__ _Float16 Bs[2][BN][LDT];
  const int tid  = threadIdx.x;
  const int wave = tid >> 5;
  const int lane = tid & 31;
  const int wm = wave & 3, wn = wave >> 2;
  const int m0 = blockIdx.x * BM, n0 = blockIdx.y * BN;

  v8f acc[2][4];
  for (int i = 0; i < 2; ++i) for (int j = 0; j < 4; ++j)
    for (int e = 0; e < 8; ++e) acc[i][j][e] = 0.f;

  // per-thread fill coordinates (4 x b128 per thread per tile: 2 for A, 2 for B)
  const int fr0 = tid >> 2,         fk0 = (tid & 3) * 8;          // chunk tid
  const int fr1 = (tid + 256) >> 2, fk1 = ((tid + 256) & 3) * 8;  // chunk tid+256
  int ga0 = m0 + fr0; if (ga0 >= M) ga0 = M - 1;                  // clamp keeps EXEC
  int ga1 = m0 + fr1; if (ga1 >= M) ga1 = M - 1;                  // full & count uniform

  const int nk = K / BK;
  // prologue: fill tile 0 into buffer 0
  async_fill_b128(A  + (size_t)ga0 * lda + fk0,        &As[0][fr0][fk0]);
  async_fill_b128(A  + (size_t)ga1 * lda + fk1,        &As[0][fr1][fk1]);
  async_fill_b128(Bt + (size_t)(n0 + fr0) * K + fk0,   &Bs[0][fr0][fk0]);
  async_fill_b128(Bt + (size_t)(n0 + fr1) * K + fk1,   &Bs[0][fr1][fk1]);

  for (int ki = 0; ki < nk; ++ki) {
    const int cur = ki & 1;
    if (ki + 1 < nk) {
      const int k1 = (ki + 1) * BK;
      async_fill_b128(A  + (size_t)ga0 * lda + k1 + fk0,      &As[cur ^ 1][fr0][fk0]);
      async_fill_b128(A  + (size_t)ga1 * lda + k1 + fk1,      &As[cur ^ 1][fr1][fk1]);
      async_fill_b128(Bt + (size_t)(n0 + fr0) * K + k1 + fk0, &Bs[cur ^ 1][fr0][fk0]);
      async_fill_b128(Bt + (size_t)(n0 + fr1) * K + k1 + fk1, &Bs[cur ^ 1][fr1][fk1]);
      if (k1 + BK < K)
        __builtin_prefetch(A + (size_t)ga0 * lda + k1 + BK + fk0, 0, 1);
      // wait for current tile's 4 fills (in-order completion; next tile's may fly)
      asm volatile("s_wait_asynccnt 0x4" ::: "memory");
    } else {
      asm volatile("s_wait_asynccnt 0x0" ::: "memory");
    }
    __syncthreads();
    v16h af[2], bf[4];
    for (int s = 0; s < 2; ++s)
      af[s] = load_frag_a(&As[cur][wm * 32 + s * 16][0], LDT, lane);
    for (int t = 0; t < 4; ++t)
      bf[t] = load_frag_b(&Bs[cur][wn * 64 + t * 16][0], LDT, lane);
    for (int sm = 0; sm < 2; ++sm)
      for (int sn = 0; sn < 4; ++sn)
        acc[sm][sn] = __builtin_amdgcn_wmma_f32_16x16x32_f16(
            false, af[sm], false, bf[sn], (short)0, acc[sm][sn], false, false);
    __syncthreads();   // protect buffer `cur` before it is refilled at ki+1
  }

  const int hi = lane >> 4, cc = lane & 15;
  for (int sm = 0; sm < 2; ++sm) for (int sn = 0; sn < 4; ++sn) {
    int rb = m0 + wm * 32 + sm * 16 + 8 * hi;
    int gc = n0 + wn * 64 + sn * 16 + cc;
    for (int e = 0; e < 8; ++e) {
      int gr = rb + e;
      if (gr >= M) continue;
      float a = acc[sm][sn][e];
      if (mode == 0) {
        a += bias[gc];
        out16[(size_t)gr * N + gc] = (_Float16)gelu_exact(a);
      } else if (mode == 1) {
        a += bias[gc];
        out32[(size_t)gr * N + gc] = a;
      } else if (mode == 3) {
        a += bias[gc];
        hrmw[(size_t)gr * N + gc] += ls[gc] * a;
      } else { // 4: qkv scatter
        int b = gr / NTOK, n = gr - b * NTOK;
        int which = gc / E, rem = gc - which * E;
        int hh = rem >> 6, dd = rem & 63;
        size_t off;
        if (which == 2)   // V stored transposed [b][h][d][n]
          off = (size_t)2 * QKVS + ((size_t)(b * NH + hh) * HD + dd) * NPAD + n;
        else              // Q, K stored [b][h][n][d]
          off = (size_t)which * QKVS + ((size_t)(b * NH + hh) * NPAD + n) * HD + dd;
        out16[off] = (_Float16)a;
      }
    }
  }
}

// ---------------- flash attention: one wave per 16-query tile ----------------
__global__ __launch_bounds__(256) void attn_wmma(
    const _Float16* __restrict__ qb, const _Float16* __restrict__ kb,
    const _Float16* __restrict__ vb,
    const float* __restrict__ temp, const float* __restrict__ locw,
    _Float16* __restrict__ o16) {
  __shared__ _Float16 pbuf[8][16][32];   // per-wave P transpose bounce
  const int wave = threadIdx.x >> 5, lane = threadIdx.x & 31;
  const int bh = blockIdx.x / 7, qc = blockIdx.x % 7;
  const int qtile = qc * 8 + wave;
  if (qtile >= QTILES) return;           // whole-wave exit, no barriers below
  const int b = bh / NH, h = bh - b * NH;
  const float scale = __expf(temp[h]);
  const float lw    = locw[h];
  const _Float16* Q = qb + (size_t)bh * NPAD * HD;
  const _Float16* K = kb + (size_t)bh * NPAD * HD;
  const _Float16* V = vb + (size_t)bh * HD * NPAD;
  const int q0 = qtile * 16;

  v16h qf[2];
  qf[0] = load_frag_a(Q + (size_t)q0 * HD,      HD, lane);
  qf[1] = load_frag_a(Q + (size_t)q0 * HD + 32, HD, lane);

  const int hi = lane >> 4, cc = lane & 15;
  int qy[8], qx[8];
  for (int e = 0; e < 8; ++e) {
    int r = q0 + e + 8 * hi; if (r >= NTOK) r = 0;
    qy[e] = r / GHN; qx[e] = r - qy[e] * GHN;
  }
  float m[8], l[8]; v8f o[4];
  for (int e = 0; e < 8; ++e) { m[e] = -INFINITY; l[e] = 0.f; }
  for (int j = 0; j < 4; ++j) for (int e = 0; e < 8; ++e) o[j][e] = 0.f;

  for (int kt = 0; kt < KTILES; ++kt) {
    const int kbase = kt * 32;
    v8f s0, s1;
    for (int e = 0; e < 8; ++e) { s0[e] = 0.f; s1[e] = 0.f; }
    for (int t = 0; t < 2; ++t) {        // two 32-wide chunks of HD=64
      v16h kf0 = load_frag_b(K + (size_t)kbase * HD + t * 32,        HD, lane);
      v16h kf1 = load_frag_b(K + (size_t)(kbase + 16) * HD + t * 32, HD, lane);
      s0 = __builtin_amdgcn_wmma_f32_16x16x32_f16(false, qf[t], false, kf0, (short)0, s0, false, false);
      s1 = __builtin_amdgcn_wmma_f32_16x16x32_f16(false, qf[t], false, kf1, (short)0, s1, false, false);
    }
    const int key0 = kbase + cc, key1 = key0 + 16;
    const int ky0 = key0 / GHN, kx0 = key0 - ky0 * GHN;
    const int ky1 = key1 / GHN, kx1 = key1 - ky1 * GHN;
    const bool ok0 = key0 < NTOK, ok1 = key1 < NTOK;
    float t0[8], t1[8], rm[8];
    for (int e = 0; e < 8; ++e) {
      int dy0 = qy[e] - ky0, dx0 = qx[e] - kx0;
      int dy1 = qy[e] - ky1, dx1 = qx[e] - kx1;
      t0[e] = ok0 ? s0[e] * scale - lw * (float)(dy0 * dy0 + dx0 * dx0) * (1.f / 1568.f) : -INFINITY;
      t1[e] = ok1 ? s1[e] * scale - lw * (float)(dy1 * dy1 + dx1 * dx1) * (1.f / 1568.f) : -INFINITY;
      rm[e] = fmaxf(t0[e], t1[e]);
    }
    for (int d = 1; d < 16; d <<= 1)
      for (int e = 0; e < 8; ++e) rm[e] = fmaxf(rm[e], __shfl_xor(rm[e], d, 32));
    float al[8], p0[8], p1[8], rs[8];
    for (int e = 0; e < 8; ++e) {
      float mn = fmaxf(m[e], rm[e]);
      al[e] = __expf(m[e] - mn);
      p0[e] = __expf(t0[e] - mn);
      p1[e] = __expf(t1[e] - mn);
      rs[e] = p0[e] + p1[e];
      m[e]  = mn;
    }
    for (int d = 1; d < 16; d <<= 1)
      for (int e = 0; e < 8; ++e) rs[e] += __shfl_xor(rs[e], d, 32);
    for (int e = 0; e < 8; ++e) l[e] = l[e] * al[e] + rs[e];
    for (int j = 0; j < 4; ++j) for (int e = 0; e < 8; ++e) o[j][e] *= al[e];
    // P (C-layout) -> LDS -> A-fragment; per-wave scratch, wave-local DS wait
    for (int e = 0; e < 8; ++e) {
      pbuf[wave][e + 8 * hi][cc]      = (_Float16)p0[e];
      pbuf[wave][e + 8 * hi][cc + 16] = (_Float16)p1[e];
    }
    asm volatile("s_wait_dscnt 0x0" ::: "memory");
    v16h pf = load_frag_a(&pbuf[wave][0][0], 32, lane);
    for (int j = 0; j < 4; ++j) {        // V^T [d][key] -> B fragments
      v16h vf = load_frag_b(V + (size_t)(j * 16) * NPAD + kbase, NPAD, lane);
      o[j] = __builtin_amdgcn_wmma_f32_16x16x32_f16(false, pf, false, vf, (short)0, o[j], false, false);
    }
  }
  for (int e = 0; e < 8; ++e) {
    int r = q0 + e + 8 * hi;
    if (r >= NTOK) continue;
    size_t row = (size_t)b * NTOK + r;
    float inv = 1.f / l[e];
    for (int j = 0; j < 4; ++j)
      o16[row * E + h * HD + j * 16 + cc] = (_Float16)(o[j][e] * inv);
  }
}

// ---------------- host orchestration ----------------
extern "C" void kernel_launch(void* const* d_in, const int* in_sizes, int n_in,
                              void* d_out, int out_size, void* d_ws, size_t ws_size,
                              hipStream_t stream) {
  (void)in_sizes; (void)n_in; (void)ws_size;
  const float* x      = (const float*)d_in[0];
  const float* tok_ng = (const float*)d_in[1];
  const float* tok_nb = (const float*)d_in[2];
  const float* tok_w1 = (const float*)d_in[3];
  const float* tok_b1 = (const float*)d_in[4];
  const float* tok_w2 = (const float*)d_in[5];
  const float* tok_b2 = (const float*)d_in[6];
  const float* ln1_g  = (const float*)d_in[7];
  const float* ln1_b  = (const float*)d_in[8];
  const float* w_qkv  = (const float*)d_in[9];
  const float* temp   = (const float*)d_in[10];
  const float* loc_w  = (const float*)d_in[11];
  const float* w_proj = (const float*)d_in[12];
  const float* b_proj = (const float*)d_in[13];
  const float* ls1    = (const float*)d_in[14];
  const float* ln2_g  = (const float*)d_in[15];
  const float* ln2_b  = (const float*)d_in[16];
  const float* ff_w1  = (const float*)d_in[17];
  const float* ff_b1  = (const float*)d_in[18];
  const float* ff_w2  = (const float*)d_in[19];
  const float* ff_b2  = (const float*)d_in[20];
  const float* ls2    = (const float*)d_in[21];

  char* ws = (char*)d_ws;
  size_t off = 0;
  auto take = [&](size_t bytes) -> char* {
    char* p = ws + off;
    off += (bytes + 255) & ~(size_t)255;
    return p;
  };
  _Float16* patches = (_Float16*)take((size_t)NT * TOKD * 2);
  _Float16* hid     = (_Float16*)take((size_t)NT * HIDD * 2);
  float*    hbuf    = (float*)   take((size_t)NT * E * 4);
  _Float16* y16     = (_Float16*)take((size_t)NT * E * 2);
  _Float16* ffh     = (_Float16*)take((size_t)NT * FFD * 2);
  _Float16* o16     = (_Float16*)take((size_t)NT * E * 2);
  _Float16* qkv     = (_Float16*)take((size_t)3 * QKVS * 2);
  _Float16* w1t     = (_Float16*)take((size_t)HIDD * TOKD * 2);
  _Float16* w2t     = (_Float16*)take((size_t)E * HIDD * 2);
  _Float16* wqkvt   = (_Float16*)take((size_t)DEPTH * 3 * E * E * 2);
  _Float16* wprojt  = (_Float16*)take((size_t)DEPTH * E * E * 2);
  _Float16* fw1t    = (_Float16*)take((size_t)DEPTH * FFD * E * 2);
  _Float16* fw2t    = (_Float16*)take((size_t)DEPTH * E * FFD * 2);

  auto tblocks = [](long n) { return (unsigned)((n + 255) / 256); };
  // weight conversion (f32 [In][Out] -> f16 [Out][In])
  wtrans_k<<<tblocks((long)TOKD * HIDD), 256, 0, stream>>>(tok_w1, w1t, TOKD, HIDD);
  wtrans_k<<<tblocks((long)HIDD * E), 256, 0, stream>>>(tok_w2, w2t, HIDD, E);
  for (int d = 0; d < DEPTH; ++d) {
    wtrans_k<<<tblocks((long)E * 3 * E), 256, 0, stream>>>(w_qkv + (size_t)d * E * 3 * E, wqkvt + (size_t)d * 3 * E * E, E, 3 * E);
    wtrans_k<<<tblocks((long)E * E), 256, 0, stream>>>(w_proj + (size_t)d * E * E, wprojt + (size_t)d * E * E, E, E);
    wtrans_k<<<tblocks((long)E * FFD), 256, 0, stream>>>(ff_w1 + (size_t)d * E * FFD, fw1t + (size_t)d * FFD * E, E, FFD);
    wtrans_k<<<tblocks((long)FFD * E), 256, 0, stream>>>(ff_w2 + (size_t)d * FFD * E, fw2t + (size_t)d * E * FFD, FFD, E);
  }

  const unsigned MT = (NT + BM - 1) / BM;  // 106
  auto gemm = [&](const _Float16* A, int lda, const _Float16* Bt, int N, int K,
                  int mode, const float* bias, const float* ls,
                  _Float16* out16p, float* out32p, float* hrmwp) {
    gemm_f16_wmma<<<dim3(MT, N / BN), 256, 0, stream>>>(
        A, lda, Bt, NT, N, K, mode, bias, ls, out16p, out32p, hrmwp);
  };

  // token embedding
  patchify_k<<<tblocks((long)NT * TOKD), 256, 0, stream>>>(x, patches, (long)NT * TOKD);
  ln_tok_k<<<NT, 256, 0, stream>>>(patches, tok_ng, tok_nb);
  gemm(patches, TOKD, w1t, HIDD, TOKD, 0, tok_b1, nullptr, hid, nullptr, nullptr);
  gemm(hid, HIDD, w2t, E, HIDD, 1, tok_b2, nullptr, nullptr, hbuf, nullptr);

  // zero q/k/v once: GEMM only writes valid rows, pad stays zero across layers
  hipMemsetAsync(qkv, 0, (size_t)3 * QKVS * 2, stream);

  for (int d = 0; d < DEPTH; ++d) {
    ln384_k<<<NT, 128, 0, stream>>>(hbuf, ln1_g + (size_t)d * E, ln1_b + (size_t)d * E, y16);
    gemm(y16, E, wqkvt + (size_t)d * 3 * E * E, 3 * E, E, 4, nullptr, nullptr, qkv, nullptr, nullptr);
    attn_wmma<<<BATCH * NH * 7, 256, 0, stream>>>(qkv, qkv + QKVS, qkv + 2 * QKVS,
                                                  temp + (size_t)d * NH, loc_w + (size_t)d * NH, o16);
    gemm(o16, E, wprojt + (size_t)d * E * E, E, E, 3,
         b_proj + (size_t)d * E, ls1 + (size_t)d * E, nullptr, nullptr, hbuf);
    ln384_k<<<NT, 128, 0, stream>>>(hbuf, ln2_g + (size_t)d * E, ln2_b + (size_t)d * E, y16);
    gemm(y16, E, fw1t + (size_t)d * FFD * E, FFD, E, 0,
         ff_b1 + (size_t)d * FFD, nullptr, ffh, nullptr, nullptr);
    gemm(ffh, FFD, fw2t + (size_t)d * E * FFD, E, FFD, 3,
         ff_b2 + (size_t)d * E, ls2 + (size_t)d * E, nullptr, nullptr, hbuf);
  }

  hipMemcpyAsync(d_out, hbuf, (size_t)out_size * sizeof(float),
                 hipMemcpyDeviceToDevice, stream);
}
